// RGAT_76072460747245
// MI455X (gfx1250) — compile-verified, matrix operand order checked by reading, not run on hardware
//
#include <hip/hip_runtime.h>
#include <hip/hip_bf16.h>

typedef __attribute__((ext_vector_type(16))) _Float16 v16h;
typedef __attribute__((ext_vector_type(8)))  _Float16 v8h;
typedef __attribute__((ext_vector_type(4)))  _Float16 v4h;
typedef __attribute__((ext_vector_type(8)))  float    v8f;
typedef int rgat_v4i __attribute__((vector_size(16)));   // matches builtin's V4i pointee

#define RGAT_N   50000
#define RGAT_E   200000
#define RGAT_R   4

#if defined(__has_builtin)
#  if __has_builtin(__builtin_amdgcn_global_load_async_to_lds_b128)
#    define USE_ASYNC_LDS 1
#  endif
#endif
#ifndef USE_ASYNC_LDS
#  define USE_ASYNC_LDS 0
#endif

__device__ __forceinline__ void wait_asynccnt0() {
#if __has_builtin(__builtin_amdgcn_s_wait_asynccnt)
    __builtin_amdgcn_s_wait_asynccnt(0);
#else
    asm volatile("s_wait_asynccnt 0x0" ::: "memory");
#endif
}

// ---------------- helpers ----------------

__device__ __forceinline__ unsigned enc_f32(float x) {
    unsigned b = __float_as_uint(x);
    return (b & 0x80000000u) ? ~b : (b | 0x80000000u);   // order-preserving encode
}
__device__ __forceinline__ float dec_f32(unsigned u) {
    return __uint_as_float((u & 0x80000000u) ? (u & 0x7FFFFFFFu) : ~u);
}

__global__ __launch_bounds__(256) void rgat_zero_f32(float* p, int n) {
    int t = blockIdx.x * 256 + threadIdx.x;
    if (t < n) p[t] = 0.0f;
}
__global__ __launch_bounds__(256) void rgat_zero_u32(unsigned* p, int n) {
    int t = blockIdx.x * 256 + threadIdx.x;
    if (t < n) p[t] = 0u;
}
__global__ __launch_bounds__(256) void rgat_cvt_f16(const float* __restrict__ in,
                                                    _Float16* __restrict__ out, int n) {
    int t = blockIdx.x * 256 + threadIdx.x;
    if (t < n) out[t] = (_Float16)in[t];
}

// ---------------- dual GEMM: fs = f16(h@Ws+bs), fd = f16(h@Wd+bd) ----------------
// h: [N][128] f16 row-major.  Ws/Wd: [128][128] f32 row-major.  bs/bd: [128] f32.
// Block = 256 threads = 8 waves; each wave owns a 16-row tile; block covers 128 rows.
// nrows must be a multiple of 16 (true: 50000) so tile validity is wave-uniform.
__global__ __launch_bounds__(256)
void rgat_gemm_dual(const _Float16* __restrict__ A,
                    const float* __restrict__ Ws, const float* __restrict__ bs,
                    const float* __restrict__ Wd, const float* __restrict__ bd,
                    _Float16* __restrict__ fs, _Float16* __restrict__ fd, int nrows)
{
    __shared__ v16h ldsB[1024];                 // 32 KB: one 128x128 f16 weight, pre-laned
#if USE_ASYNC_LDS
    __shared__ float ldsRaw[128 * 128];         // 64 KB raw f32 staging (async copy target)
#endif
    _Float16* lptr = (_Float16*)ldsB;

    const int tid  = threadIdx.x;
    const int lane = tid & 31;
    const int wave = tid >> 5;

    const int  row0      = (blockIdx.x * 8 + wave) * 16;
    const bool tileValid = (row0 < nrows);      // uniform per wave

    // ---- load A fragments (4 k-chunks of 32) once; shared by both GEMMs ----
    const int r = tileValid ? (row0 + (lane & 15)) : 0;
    const _Float16* arow = A + (long)r * 128;
    const int aoff = (lane < 16) ? 0 : 8;       // wave32 f16 A-matrix lane layout

    v16h afrag[4];
#pragma unroll
    for (int kc = 0; kc < 4; ++kc) {
        v8h lo = *(const v8h*)(arow + kc * 32 + aoff);       // K = base+{0..7}  (or +8..15)
        v8h hi = *(const v8h*)(arow + kc * 32 + aoff + 16);  // K = base+{16..23}(or +24..31)
#pragma unroll
        for (int t = 0; t < 8; ++t) { afrag[kc][t] = lo[t]; afrag[kc][8 + t] = hi[t]; }
    }

    for (int m = 0; m < 2; ++m) {
        const float* W    = m ? Wd : Ws;
        const float* bias = m ? bd : bs;
        _Float16*    out  = m ? fd : fs;

        __syncthreads();                        // prior-iteration LDS reads complete

#if USE_ASYNC_LDS
        // ---- async stage: 64 KB f32 weight, global -> LDS, coalesced 16B/lane chunks ----
        {
            const float* gsrc = W + tid * 64;   // 256 threads * 64 floats = 16384
            float*       ldst = ldsRaw + tid * 64;
#pragma unroll
            for (int c = 0; c < 16; ++c) {
                __builtin_amdgcn_global_load_async_to_lds_b128(
                    (__attribute__((address_space(1))) rgat_v4i*)(uintptr_t)(gsrc + c * 4),
                    (__attribute__((address_space(3))) rgat_v4i*)(ldst + c * 4),
                    0, 0);
            }
            wait_asynccnt0();
        }
        __syncthreads();                        // ldsRaw fully populated
#endif

        // ---- pack into WMMA-B lane layout (source-linear => coalesced reads) ----
        for (int i = tid; i < 128 * 128; i += 256) {
            int K  = i >> 7;                    // 0..127
            int Nc = i & 127;                   // 0..127
            int kc = K >> 5, kl = K & 31;
            int nt = Nc >> 4, nl = Nc & 15;
            int ln, j;
            if      (kl < 8)  { ln = nl;      j = kl;      }
            else if (kl < 16) { ln = nl + 16; j = kl - 8;  }
            else if (kl < 24) { ln = nl;      j = kl - 8;  }
            else              { ln = nl + 16; j = kl - 16; }
#if USE_ASYNC_LDS
            float v = ldsRaw[i];
#else
            float v = W[i];
#endif
            lptr[((nt * 4 + kc) * 32 + ln) * 16 + j] = (_Float16)v;
        }
        __syncthreads();

        if (tileValid) {                        // wave-uniform; EXEC all-1s inside
#pragma unroll
            for (int nt = 0; nt < 8; ++nt) {
                v8f acc = {};
#pragma unroll
                for (int kc = 0; kc < 4; ++kc) {
                    v16h bfrag = ldsB[(nt * 4 + kc) * 32 + lane];
                    acc = __builtin_amdgcn_wmma_f32_16x16x32_f16(
                        /*neg_a=*/false, afrag[kc], /*neg_b=*/false, bfrag,
                        /*c_mod=*/(short)0, acc, /*reuse_a=*/false, /*reuse_b=*/false);
                }
                const int   col   = nt * 16 + (lane & 15);
                const float b     = bias[col];
                const int   rbase = row0 + ((lane < 16) ? 0 : 8);
#pragma unroll
                for (int i2 = 0; i2 < 8; ++i2)
                    out[(long)(rbase + i2) * 128 + col] = (_Float16)(acc[i2] + b);
            }
        }
    }
}

// ---------------- edge pass 1: score + segment max ----------------
// one wave per edge; lane owns hd = lane*4..lane*4+3 (head = lane>>3)
__global__ __launch_bounds__(256)
void rgat_edge_score(const _Float16* __restrict__ fs, const _Float16* __restrict__ fd,
                     const int* __restrict__ src, const int* __restrict__ dst,
                     const float* __restrict__ att,
                     float* __restrict__ score, unsigned* __restrict__ nodemax, int nE)
{
    int e    = blockIdx.x * 8 + (threadIdx.x >> 5);
    int lane = threadIdx.x & 31;
    if (e >= nE) return;
    int s = src[e], d = dst[e];

    v4h a = *(const v4h*)(fs + (long)s * 128 + lane * 4);
    v4h b = *(const v4h*)(fd + (long)d * 128 + lane * 4);

    float partial = 0.0f;
#pragma unroll
    for (int j = 0; j < 4; ++j) {
        float v = (float)a[j] + (float)b[j];
        v = (v > 0.0f) ? v : 0.2f * v;          // leaky_relu(.,0.2)
        partial += v * att[lane * 4 + j];
    }
    // reduce 8-lane head groups (heads are 8-lane aligned)
    partial += __shfl_xor(partial, 1, 32);
    partial += __shfl_xor(partial, 2, 32);
    partial += __shfl_xor(partial, 4, 32);

    if ((lane & 7) == 0) {
        int h = lane >> 3;
        score[(long)e * 4 + h] = partial;
        atomicMax(&nodemax[d * 4 + h], enc_f32(partial));
    }
}

// ---------------- edge pass 2: exp + segment-sum denominator ----------------
__global__ __launch_bounds__(256)
void rgat_edge_softmax(const int* __restrict__ dst, float* __restrict__ score,
                       const unsigned* __restrict__ nodemax, float* __restrict__ den, int nEH)
{
    int t = blockIdx.x * 256 + threadIdx.x;
    if (t >= nEH) return;
    int e = t >> 2, h = t & 3;
    int d = dst[e];
    float m = dec_f32(nodemax[d * 4 + h]);
    if (!isfinite(m)) m = 0.0f;                 // zero-in-degree guard
    float ex = expf(score[t] - m);
    score[t] = ex;
    atomicAdd(&den[d * 4 + h], ex);
}

// ---------------- edge pass 3: normalize + scatter-sum ----------------
__global__ __launch_bounds__(256)
void rgat_edge_scatter(const _Float16* __restrict__ fs,
                       const int* __restrict__ src, const int* __restrict__ dst,
                       const float* __restrict__ score, const float* __restrict__ den,
                       float* __restrict__ acc, int nE)
{
    int e    = blockIdx.x * 8 + (threadIdx.x >> 5);
    int lane = threadIdx.x & 31;
    if (e >= nE) return;
    int s = src[e], d = dst[e];
    int h = lane >> 3;
    float a = score[(long)e * 4 + h] / fmaxf(den[d * 4 + h], 1e-16f);

    v4h fv = *(const v4h*)(fs + (long)s * 128 + lane * 4);
    float* pd = acc + (long)d * 128 + lane * 4;
#pragma unroll
    for (int j = 0; j < 4; ++j) atomicAdd(pd + j, a * (float)fv[j]);
}

// ---------------- final layer: mean over 4 heads ----------------
__global__ __launch_bounds__(256)
void rgat_head_mean(const float* __restrict__ acc, float* __restrict__ out, int nOut)
{
    int t = blockIdx.x * 256 + threadIdx.x;
    if (t >= nOut) return;
    int dcol = t & 31;
    long node = t >> 5;
    const float* p = acc + node * 128 + dcol;
    out[t] = 0.25f * (p[0] + p[32] + p[64] + p[96]);
}

// ---------------- host ----------------

extern "C" void kernel_launch(void* const* d_in, const int* in_sizes, int n_in,
                              void* d_out, int out_size, void* d_ws, size_t ws_size,
                              hipStream_t stream)
{
    (void)in_sizes; (void)n_in; (void)out_size; (void)ws_size;
    const int N = RGAT_N, E = RGAT_E, R = RGAT_R;

    const float* x    = (const float*)d_in[0];
    const int*   esrc = (const int*)d_in[1];
    const int*   edst = (const int*)d_in[2];
    const float *W[3][2], *B[3][2], *ATT[3];
    for (int l = 0; l < 3; ++l) {
        int b = 3 + 5 * l;
        W[l][0] = (const float*)d_in[b + 0];  B[l][0] = (const float*)d_in[b + 1];
        W[l][1] = (const float*)d_in[b + 2];  B[l][1] = (const float*)d_in[b + 3];
        ATT[l]  = (const float*)d_in[b + 4];
    }

    // workspace carve-out (256-B aligned)
    char* ws = (char*)d_ws; size_t o = 0;
    auto take = [&](size_t bytes) -> char* {
        char* p = ws + o; o = (o + bytes + 255) & ~(size_t)255; return p;
    };
    _Float16* hf16    = (_Float16*)take((size_t)N * 128 * 2);
    _Float16* fs16    = (_Float16*)take((size_t)N * 128 * 2);
    _Float16* fd16    = (_Float16*)take((size_t)N * 128 * 2);
    float*    acc     = (float*)   take((size_t)N * 128 * 4);
    float*    score   = (float*)   take((size_t)E * 4 * 4);
    unsigned* nodemax = (unsigned*)take((size_t)N * 4 * 4);
    float*    den     = (float*)   take((size_t)N * 4 * 4);

    const int gCvtH   = (N * 128 + 255) / 256;
    const int gGemm   = (N + 127) / 128;
    const int gEdgeW  = (E + 7) / 8;          // wave-per-edge kernels
    const int gEdgeEH = (E * 4 + 255) / 256;
    const int gNH     = (N * 4 + 255) / 256;
    const int gOut    = (N * 32 + 255) / 256;

    rgat_cvt_f16<<<gCvtH, 256, 0, stream>>>(x, hf16, N * 128);

    for (int l = 0; l < 3; ++l) {
        rgat_zero_f32<<<gCvtH, 256, 0, stream>>>(acc, N * 128);
        for (int r = 0; r < R; ++r) {
            rgat_gemm_dual<<<gGemm, 256, 0, stream>>>(
                hf16,
                W[l][0] + (size_t)r * 128 * 128, B[l][0] + (size_t)r * 128,
                W[l][1] + (size_t)r * 128 * 128, B[l][1] + (size_t)r * 128,
                fs16, fd16, N);

            rgat_zero_u32<<<gNH, 256, 0, stream>>>(nodemax, N * 4);
            rgat_zero_f32<<<gNH, 256, 0, stream>>>(den, N * 4);

            rgat_edge_score<<<gEdgeW, 256, 0, stream>>>(
                fs16, fd16, esrc + (size_t)r * E, edst + (size_t)r * E,
                ATT[l] + (size_t)r * 128, score, nodemax, E);
            rgat_edge_softmax<<<gEdgeEH, 256, 0, stream>>>(
                edst + (size_t)r * E, score, nodemax, den, E * 4);
            rgat_edge_scatter<<<gEdgeW, 256, 0, stream>>>(
                fs16, esrc + (size_t)r * E, edst + (size_t)r * E,
                score, den, acc, E);
        }
        if (l < 2) {
            rgat_cvt_f16<<<gCvtH, 256, 0, stream>>>(acc, hf16, N * 128);  // flatten == identity
        } else {
            rgat_head_mean<<<gOut, 256, 0, stream>>>(acc, (float*)d_out, N * 32);
        }
    }
}
